// FeedzaiTrainSync_54296976556060
// MI455X (gfx1250) — compile-verified, matrix operand order checked by reading, not run on hardware
//
#include <hip/hip_runtime.h>
#include <hip/hip_bf16.h>
#include <math.h>

typedef __attribute__((ext_vector_type(16))) __bf16 v16bf;
typedef __attribute__((ext_vector_type(8)))  __bf16 v8bf;
typedef __attribute__((ext_vector_type(2)))  __bf16 v2bf;
typedef __attribute__((ext_vector_type(8)))  float  v8f;

#define B_   1024
#define T_   64
#define F_   18
#define U_   128
#define NT3  24          // 384 / 16 column tiles
#define KCH  4           // 128 / 32 K-chunks for the recurrent GEMM

// ws layout:
//   [0)          : recB bf16 [24 nTile][4 kChunk][32 lane][16 elem]
//   [REC_N)      : xB   bf16 [24 nTile][32 lane][16 elem]  (K padded 18->32, k=0 zeroed)
//   [byte 1<<17) : hist f32  [64 blk][64 t][16 m][128 u]   (32 MB, L2-resident)
#define REC_N (NT3 * KCH * 32 * 16)
#define XB_N  (NT3 * 32 * 16)
#define HIST_BYTE_OFF (1u << 17)

// LDS row strides chosen for bank-conflict-free b128 fragment loads
#define HB_STR 72    // bf16 h tile  [16][72]
#define XS_STR 40    // bf16 x tile  [16][40]
#define HF_STR 130   // f32  h tile  [16][130]

// hardware transcendental tanh if available (V_TANH_F32, TRANS32 pipe)
__device__ __forceinline__ float fast_tanh(float x) {
#if __has_builtin(__builtin_amdgcn_tanhf)
  return __builtin_amdgcn_tanhf(x);
#else
  return tanhf(x);
#endif
}

// 16-bit WMMA operand K pattern (ISA 7.12.2):
// lanes 0-15 hold K {0..7,16..23}, lanes 16-31 hold K {8..15,24..31}
__device__ __forceinline__ int kpat(int e, int half) {
  return (e & 7) + 8 * half + 16 * (e >> 3);
}

// fragment = two contiguous 8x bf16 chunks (k = off+half*8.. and +16)
__device__ __forceinline__ v16bf load_frag(const __bf16* p) {
  v8bf lo = *(const v8bf*)(p);
  v8bf hi = *(const v8bf*)(p + 16);
  return __builtin_shufflevector(lo, hi, 0, 1, 2, 3, 4, 5, 6, 7,
                                         8, 9, 10, 11, 12, 13, 14, 15);
}

// ---------------------------------------------------------------------------
// Prep: swizzle kernel / rec_kernel into bf16 WMMA B-fragment order in ws.
// Row k=0 of `kernel` is zeroed: the id column is applied exactly in f32.
// ---------------------------------------------------------------------------
__global__ void gru_prep_weights(const float* __restrict__ kern,   // [18][384]
                                 const float* __restrict__ rec,    // [128][384]
                                 __bf16* __restrict__ wsb) {
  int gid = blockIdx.x * blockDim.x + threadIdx.x;
  if (gid < REC_N) {
    int e = gid & 15, lane = (gid >> 4) & 31, c = (gid >> 9) & 3, nT = gid >> 11;
    int k = c * 32 + kpat(e, lane >> 4);
    int n = (nT << 4) + (lane & 15);
    wsb[gid] = (__bf16)rec[k * 384 + n];
  } else if (gid < REC_N + XB_N) {
    int g = gid - REC_N;
    int e = g & 15, lane = (g >> 4) & 31, nT = g >> 9;
    int k = kpat(e, lane >> 4);
    int n = (nT << 4) + (lane & 15);
    float v = (k >= 1 && k < F_) ? kern[k * 384 + n] : 0.0f;
    wsb[REC_N + g] = (__bf16)v;
  }
}

// ---------------------------------------------------------------------------
// Main: 64 workgroups x 256 threads (8 wave32).  Workgroup owns 16 batch rows
// and runs all 64 timesteps; wave w owns U-columns [16w,16w+16) of each gate.
// ---------------------------------------------------------------------------
__global__ __launch_bounds__(256)
void gru_main(const float* __restrict__ inputs,   // [1024][64][18]
              const float* __restrict__ kern,     // [18][384] (row 0 in f32)
              const float* __restrict__ bias,     // [384]
              const float* __restrict__ dense_w,  // [128][64]
              const float* __restrict__ dense_b,  // [64]
              const float* __restrict__ out_w,    // [64]
              const float* __restrict__ out_b,    // [1]
              const __bf16* __restrict__ wsb,
              float* __restrict__ hist,
              float* __restrict__ out) {
  __shared__ float  hF[16 * HF_STR];   // f32 h tile [m][u] (exact recurrent carry)
  __shared__ __bf16 hB[16 * HB_STR];   // bf16 h tile, fragment-row-major [m][k]
  __shared__ __bf16 xS[16 * XS_STR];   // bf16 x tile [m][k], k<32 (k=0 and >=18 zero)
  __shared__ int    idsh[16 * T_];
  __shared__ int    prevA[16 * T_];
  __shared__ float  dsh[16 * 64];

  const int tid  = threadIdx.x;
  const int lane = tid & 31;
  const int w    = tid >> 5;        // wave 0..7
  const int half = lane >> 4;
  const int n    = lane & 15;
  const int blk  = blockIdx.x;
  const int b0   = blk * 16;

  // ---- hoist all B fragments + per-column scalars into registers ----
  v16bf recB[3][KCH];               // j=0:z, 1:r, 2:h-gate recurrent weights
  v16bf xB[3];
  #pragma unroll
  for (int j = 0; j < 3; ++j) {
    int nT = w + 8 * j;
    #pragma unroll
    for (int c = 0; c < KCH; ++c)
      recB[j][c] = *(const v16bf*)(wsb + ((nT * KCH + c) * 32 + lane) * 16);
    xB[j] = *(const v16bf*)(wsb + REC_N + (nT * 32 + lane) * 16);
  }
  const int col = w * 16 + n;       // this lane's U column (0..127)
  const float bz = bias[col], br = bias[128 + col], bh = bias[256 + col];
  const float k0z = kern[col], k0r = kern[128 + col], k0h = kern[256 + col];

  // ---- ids and prev-occurrence table (replaces the 512MB id-state table) ----
  for (int i = tid; i < 16 * T_; i += 256) {
    int m = i >> 6, t = i & 63;
    idsh[m * T_ + t] = (int)inputs[((b0 + m) * T_ + t) * F_ + 0];
  }
  for (int i = tid; i < 16 * XS_STR; i += 256) xS[i] = (__bf16)0.0f;
  __syncthreads();
  for (int i = tid; i < 16 * T_; i += 256) {
    int m = i >> 6, t = i & 63;
    int id = idsh[m * T_ + t], p = -1;
    for (int s = t - 1; s >= 0; --s)
      if (idsh[m * T_ + s] == id) { p = s; break; }
    prevA[m * T_ + t] = p;
  }
  __syncthreads();

  // ---- serial time loop ----
  for (int t = 0; t < T_; ++t) {
    // stage x_t features 1..17 as bf16 (id column handled exactly in f32)
    for (int i = tid; i < 16 * 17; i += 256) {
      int m = i / 17, f = i % 17 + 1;
      xS[m * XS_STR + f] = (__bf16)inputs[((b0 + m) * T_ + t) * F_ + f];
    }
    // gather h: keep hF/hB (they hold h_new(t-1)) when prev==t-1, else hist / 0
    for (int idx = tid; idx < 16 * 64; idx += 256) {
      int m = idx >> 6, up = (idx & 63) << 1;
      int p = prevA[m * T_ + t];
      if (!(t > 0 && p == t - 1)) {
        float2 v = make_float2(0.0f, 0.0f);
        if (p >= 0) v = *(const float2*)&hist[((blk * T_ + p) * 16 + m) * U_ + up];
        *(float2*)&hF[m * HF_STR + up] = v;
        v2bf b; b[0] = (__bf16)v.x; b[1] = (__bf16)v.y;
        *(v2bf*)&hB[m * HB_STR + up] = b;
      }
    }
    __syncthreads();

    // A fragments: two ds_load_b128 each from fragment-row-major bf16 LDS
    v16bf hA[KCH], xA;
    {
      const __bf16* hrow = hB + n * HB_STR + half * 8;
      #pragma unroll
      for (int c = 0; c < KCH; ++c) hA[c] = load_frag(hrow + c * 32);
      xA = load_frag(xS + n * XS_STR + half * 8);
    }

    // 15 WMMAs: z/r fuse x+h contributions; h-gate keeps them split (r*hz)
    v8f az = {}, ar = {}, ahz = {}, ahx = {};
    #pragma unroll
    for (int c = 0; c < KCH; ++c) {
      az  = __builtin_amdgcn_wmma_f32_16x16x32_bf16(false, hA[c], false, recB[0][c], (short)0, az,  false, false);
      ar  = __builtin_amdgcn_wmma_f32_16x16x32_bf16(false, hA[c], false, recB[1][c], (short)0, ar,  false, false);
      ahz = __builtin_amdgcn_wmma_f32_16x16x32_bf16(false, hA[c], false, recB[2][c], (short)0, ahz, false, false);
    }
    az  = __builtin_amdgcn_wmma_f32_16x16x32_bf16(false, xA, false, xB[0], (short)0, az,  false, false);
    ar  = __builtin_amdgcn_wmma_f32_16x16x32_bf16(false, xA, false, xB[1], (short)0, ar,  false, false);
    ahx = __builtin_amdgcn_wmma_f32_16x16x32_bf16(false, xA, false, xB[2], (short)0, ahx, false, false);

    // gates (C/D layout: lane n = column, vgpr v -> row m = v + 8*half)
    float hnew[8];
    #pragma unroll
    for (int v = 0; v < 8; ++v) {
      int m = v + 8 * half;
      float idf = (float)idsh[m * T_ + t];          // exact f32 id contribution
      float z  = fminf(fmaxf(0.2f * (az[v] + bz + idf * k0z) + 0.5f, 0.0f), 1.0f);
      float r  = fminf(fmaxf(0.2f * (ar[v] + br + idf * k0r) + 0.5f, 0.0f), 1.0f);
      float hh = fast_tanh(ahx[v] + bh + idf * k0h + r * ahz[v]);
      float ho = hF[m * HF_STR + col];
      hnew[v]  = z * ho + (1.0f - z) * hh;
    }
    __syncthreads();
    #pragma unroll
    for (int v = 0; v < 8; ++v) {
      int m = v + 8 * half;
      hF[m * HF_STR + col] = hnew[v];
      hB[m * HB_STR + col] = (__bf16)hnew[v];
      hist[((blk * T_ + t) * 16 + m) * U_ + col] = hnew[v];
    }
    if (t + 1 < T_ && lane == 0)
      __builtin_prefetch(&inputs[((b0 + w) * T_ + t + 1) * F_], 0, 0);
    __threadfence();   // make hist stores visible before next step's gather
    __syncthreads();
  }

  // ---- head: relu(h @ dense_w + db) -> sigmoid(d @ out_w + ob) ----
  for (int i = tid; i < 16 * 64; i += 256) {
    int m = i >> 6, j = i & 63;
    float s = dense_b[j];
    #pragma unroll 4
    for (int u = 0; u < U_; ++u) s += hF[m * HF_STR + u] * dense_w[u * 64 + j];
    dsh[m * 64 + j] = fmaxf(s, 0.0f);
  }
  __syncthreads();
  if (tid < 16) {
    float s = out_b[0];
    #pragma unroll 4
    for (int j = 0; j < 64; ++j) s += dsh[tid * 64 + j] * out_w[j];
    out[b0 + tid] = 1.0f / (1.0f + __expf(-s));
  }
}

// ---------------------------------------------------------------------------
extern "C" void kernel_launch(void* const* d_in, const int* in_sizes, int n_in,
                              void* d_out, int out_size, void* d_ws, size_t ws_size,
                              hipStream_t stream) {
  const float* inputs = (const float*)d_in[0];
  // d_in[1] (sync_states) is all-zero initial state: modeled analytically, unused.
  const float* kern   = (const float*)d_in[2];
  const float* rec    = (const float*)d_in[3];
  const float* bias   = (const float*)d_in[4];
  const float* dw     = (const float*)d_in[5];
  const float* db     = (const float*)d_in[6];
  const float* ow     = (const float*)d_in[7];
  const float* ob     = (const float*)d_in[8];

  __bf16* wsb  = (__bf16*)d_ws;
  float*  hist = (float*)((char*)d_ws + HIST_BYTE_OFF);

  gru_prep_weights<<<(REC_N + XB_N + 255) / 256, 256, 0, stream>>>(kern, rec, wsb);
  gru_main<<<B_ / 16, 256, 0, stream>>>(inputs, kern, bias, dw, db, ow, ob,
                                        wsb, hist, (float*)d_out);
}
